// RelativeMultiheadAttention_10264971837453
// MI455X (gfx1250) — matrix operand order repaired and checked by reading, not compiled, not run on hardware
//
#include <hip/hip_runtime.h>

typedef __bf16 bf16;
typedef __attribute__((ext_vector_type(4)))  __bf16 v4bf;
typedef __attribute__((ext_vector_type(16))) __bf16 v16bf;
typedef __attribute__((ext_vector_type(8)))  float  v8f;

__device__ __forceinline__ v8f wmma_bf16(v16bf a, v16bf b, v8f c) {
  // D = A(16x32 bf16) * B(32x16 bf16) + C(16x16 f32)
  return __builtin_amdgcn_wmma_f32_16x16x32_bf16(false, a, false, b, (short)0, c,
                                                 false, false);
}

__device__ __forceinline__ v4bf cvt4(float4 f) {
  v4bf v = { (bf16)f.x, (bf16)f.y, (bf16)f.z, (bf16)f.w };
  return v;
}

// A fragment: rows row0..row0+15, K-slice ks0..ks0+31 from row-major LDS [row][k].
// Per ISA 7.12.2 (16-bit A 16x32): lane L -> M=L%16, half=L/16;
// regs 0..3 hold K = half*8 + {0..7}, regs 4..7 hold K = 16 + half*8 + {0..7}.
__device__ __forceinline__ v16bf ldA(const bf16* S, int stride, int row0, int ks0, int lane) {
  int r = row0 + (lane & 15);
  int half = (lane >> 4) & 1;
  const bf16* p = S + r * stride + ks0;
  v16bf v;
#pragma unroll
  for (int t = 0; t < 8; ++t) {
    v[t]     = p[half * 8 + t];
    v[t + 8] = p[16 + half * 8 + t];
  }
  return v;
}

// B fragment: cols col0..col0+15, K-slice ks0..ks0+31; LDS holds B as [n][k].
__device__ __forceinline__ v16bf ldB(const bf16* S, int stride, int col0, int ks0, int lane) {
  int r = col0 + (lane & 15);
  int half = (lane >> 4) & 1;
  const bf16* p = S + r * stride + ks0;
  v16bf v;
#pragma unroll
  for (int t = 0; t < 8; ++t) {
    v[t]     = p[half * 8 + t];
    v[t + 8] = p[16 + half * 8 + t];
  }
  return v;
}

// -------------------- C[M,N] = A[M,K] @ B[N,K]^T  (f32 in/out, bf16 WMMA) ----
template <bool GUARD>
__global__ __launch_bounds__(256) void gemm_bt_kernel(
    const float* __restrict__ A, const float* __restrict__ B, float* __restrict__ C,
    int M, int N, int K) {
  __shared__ bf16 As[128 * 36];
  __shared__ bf16 Bs[128 * 36];
  int tid = threadIdx.x;
  int lane = tid & 31, wave = tid >> 5;
  int wr = wave & 3, wc = wave >> 2;            // wave tile: rows 32*wr, cols 64*wc
  int bm = blockIdx.y * 128, bn = blockIdx.x * 128;
  v8f zero = {};
  v8f acc[2][4];
#pragma unroll
  for (int i = 0; i < 2; ++i)
#pragma unroll
    for (int j = 0; j < 4; ++j) acc[i][j] = zero;

  for (int k0 = 0; k0 < K; k0 += 32) {
    __syncthreads();
    // stage 128x32 tiles as float4 -> packed bf16x4 (1024 quads, 4 per thread)
#pragma unroll
    for (int s = 0; s < 4; ++s) {
      int idx = s * 256 + tid;
      int r = idx >> 3;                         // 0..127
      int c = (idx & 7) << 2;                   // 0,4,..,28
      float4 av = {0.f, 0.f, 0.f, 0.f};
      if (!GUARD || (bm + r < M))
        av = *(const float4*)&A[(size_t)(bm + r) * K + (k0 + c)];
      *(v4bf*)&As[r * 36 + c] = cvt4(av);
      float4 bv = {0.f, 0.f, 0.f, 0.f};
      if (!GUARD || (bn + r < N))
        bv = *(const float4*)&B[(size_t)(bn + r) * K + (k0 + c)];
      *(v4bf*)&Bs[r * 36 + c] = cvt4(bv);
    }
    // prefetch next K-tile while this one is consumed (global_prefetch_b8)
    if (k0 + 32 < K) {
      int pr = tid >> 1;
      int pc = (tid & 1) * 16;
      int ra = GUARD ? min(bm + pr, M - 1) : (bm + pr);
      int rb = GUARD ? min(bn + pr, N - 1) : (bn + pr);
      __builtin_prefetch(&A[(size_t)ra * K + (k0 + 32 + pc)], 0, 1);
      __builtin_prefetch(&B[(size_t)rb * K + (k0 + 32 + pc)], 0, 1);
    }
    __syncthreads();
#pragma unroll
    for (int rt = 0; rt < 2; ++rt) {
      v16bf a = ldA(As, 36, 32 * wr + 16 * rt, 0, lane);
#pragma unroll
      for (int ct = 0; ct < 4; ++ct) {
        v16bf b = ldB(Bs, 36, 64 * wc + 16 * ct, 0, lane);
        acc[rt][ct] = wmma_bf16(a, b, acc[rt][ct]);
      }
    }
  }
  int ln = lane & 15, lh = lane >> 4;           // D: lane->N=L%16, reg g->M=g+8*(L/16)
#pragma unroll
  for (int rt = 0; rt < 2; ++rt)
#pragma unroll
    for (int ct = 0; ct < 4; ++ct)
#pragma unroll
      for (int g = 0; g < 8; ++g) {
        int row = bm + 32 * wr + 16 * rt + g + lh * 8;
        int col = bn + 64 * wc + 16 * ct + ln;
        if (!GUARD || row < M) C[(size_t)row * N + col] = acc[rt][ct][g];
      }
}

// -------------------- bias row-vectors -------------------------------------
// cb[b=(n,h)][j] = sum_e qc[h*64+e] * Kp[j, n*1024 + h*64 + e]
__global__ void cbias_kernel(const float* __restrict__ Kp, const float* __restrict__ qc,
                             float* __restrict__ cb) {
  int idx = blockIdx.x * 256 + threadIdx.x;
  if (idx >= 128 * 512) return;
  int b = idx >> 9, j = idx & 511;
  int n = b >> 4, h = b & 15;
  const float* kp = Kp + (size_t)j * 8192 + (size_t)n * 1024 + h * 64;
  const float* qv = qc + h * 64;
  float s = 0.0f;
#pragma unroll 8
  for (int e = 0; e < 64; ++e) s += qv[e] * kp[e];
  cb[idx] = s;
}

// ub[b=(n,h)][rr] = sum_e ql[h*64+e] * LKp[(rr*8+n)*1024 + h*64 + e]
__global__ void ubias_kernel(const float* __restrict__ LKp, const float* __restrict__ ql,
                             float* __restrict__ ub) {
  int idx = blockIdx.x * 256 + threadIdx.x;
  if (idx >= 128 * 1025) return;
  int b = idx / 1025, rr = idx % 1025;
  int n = b >> 4, h = b & 15;
  const float* lp = LKp + (size_t)(rr * 8 + n) * 1024 + h * 64;
  const float* qv = ql + h * 64;
  float s = 0.0f;
#pragma unroll 8
  for (int e = 0; e < 64; ++e) s += qv[e] * lp[e];
  ub[idx] = s;
}

// -------------------- fused relative attention -----------------------------
// grid (128 batch-heads, 8 q-tiles), block 256 (8 waves).
// skew(unskewed, 512) reduces to skewed[i,j] = U[i, j-i+63] over the band
// U = q @ lk2[:, rr0..rr0+127], rr0 = 512+j0-i0-63 (tri-masks provably zero).
__global__ __launch_bounds__(256) void attn_kernel(
    const float* __restrict__ Qp, const float* __restrict__ Kp, const float* __restrict__ Vp,
    const float* __restrict__ LKp, const float* __restrict__ cb, const float* __restrict__ ub,
    const float* __restrict__ mask, float* __restrict__ O) {
  __shared__ bf16 ks[64 * 64];     // [j][e]
  __shared__ bf16 vs[64 * 64];     // [e][j]  (transposed stage)
  __shared__ bf16 lks[128 * 64];   // [rr_local][e]
  __shared__ float sS[64 * 64];    // logits tile
  __shared__ bf16 Ps[64 * 64];     // softmax probs (also Q staging before loop)
  __shared__ float red[64 * 4];
  __shared__ float mrow[64], lrow[64], arow[64];

  int b = blockIdx.x;
  int i0 = blockIdx.y * 64;
  int n = b >> 4, h = b & 15;
  int tid = threadIdx.x, lane = tid & 31, wave = tid >> 5;
  int wr = wave & 3, wc = wave >> 2;
  int ln = lane & 15, lh = lane >> 4;
  size_t headoff = (size_t)n * 1024 + (size_t)h * 64;

  for (int s = tid; s < 1024; s += 256) {        // stage Q into Ps (temp), float4
    int r = s >> 4, c = (s & 15) << 2;
    float4 qv = *(const float4*)&Qp[(size_t)(i0 + r) * 8192 + headoff + c];
    *(v4bf*)&Ps[r * 64 + c] = cvt4(qv);
  }
  if (tid < 64) { mrow[tid] = -3.0e38f; lrow[tid] = 0.0f; }
  __syncthreads();
  v16bf qa0 = ldA(Ps, 64, 16 * wr, 0, lane);     // Q frags kept in VGPRs
  v16bf qa1 = ldA(Ps, 64, 16 * wr, 32, lane);

  v8f zero = {};
  v8f oacc[2]; oacc[0] = zero; oacc[1] = zero;   // rows 16*wr, cols 32*wc of O

  for (int j0 = 0; j0 < 512; j0 += 64) {
    __syncthreads();
    for (int s = tid; s < 1024; s += 256) {
      int r = s >> 4, c = (s & 15) << 2;
      float4 kv = *(const float4*)&Kp[(size_t)(j0 + r) * 8192 + headoff + c];
      *(v4bf*)&ks[r * 64 + c] = cvt4(kv);
      float4 vv = *(const float4*)&Vp[(size_t)(j0 + r) * 8192 + headoff + c];
      vs[(c + 0) * 64 + r] = (bf16)vv.x;         // transpose to [e][j]
      vs[(c + 1) * 64 + r] = (bf16)vv.y;
      vs[(c + 2) * 64 + r] = (bf16)vv.z;
      vs[(c + 3) * 64 + r] = (bf16)vv.w;
    }
    int rr0 = 512 + j0 - i0 - 63;                // rr0+r always in [1,1024]: no guard
    for (int s = tid; s < 2048; s += 256) {
      int r = s >> 4, c = (s & 15) << 2;
      int rr = rr0 + r;
      float4 lv = *(const float4*)&LKp[(size_t)(rr * 8 + n) * 1024 + (size_t)h * 64 + c];
      *(v4bf*)&lks[r * 64 + c] = cvt4(lv);
    }
    __syncthreads();

    // ---- content logits: Q (16x64) @ K^T (64 x 32wc slice) ----
    v8f sacc[2]; sacc[0] = zero; sacc[1] = zero;
#pragma unroll
    for (int ct = 0; ct < 2; ++ct) {
      v16bf b0 = ldB(ks, 64, 32 * wc + 16 * ct, 0, lane);
      v16bf b1 = ldB(ks, 64, 32 * wc + 16 * ct, 32, lane);
      sacc[ct] = wmma_bf16(qa0, b0, sacc[ct]);
      sacc[ct] = wmma_bf16(qa1, b1, sacc[ct]);
    }
#pragma unroll
    for (int ct = 0; ct < 2; ++ct)
#pragma unroll
      for (int g = 0; g < 8; ++g) {
        int rl = 16 * wr + g + lh * 8;
        int cl = 32 * wc + 16 * ct + ln;
        int j = j0 + cl;
        sS[rl * 64 + cl] = sacc[ct][g] + cb[b * 512 + j]
                         + mask[(size_t)n * 262144 + (size_t)(i0 + rl) * 512 + j];
      }
    __syncthreads();

    // ---- relative logits: U = Q @ lk-band, skew-gathered into sS ----
#pragma unroll
    for (int ct = 0; ct < 4; ++ct) {
      v16bf b0 = ldB(lks, 64, 64 * wc + 16 * ct, 0, lane);
      v16bf b1 = ldB(lks, 64, 64 * wc + 16 * ct, 32, lane);
      v8f ua = zero;
      ua = wmma_bf16(qa0, b0, ua);
      ua = wmma_bf16(qa1, b1, ua);
#pragma unroll
      for (int g = 0; g < 8; ++g) {
        int rl = 16 * wr + g + lh * 8;
        int uc = 64 * wc + 16 * ct + ln;
        int jl = uc - 63 + rl;                   // skew: band col rr0+uc -> key j0+jl
        if (jl >= 0 && jl < 64)
          sS[rl * 64 + jl] += ua[g] + ub[b * 1025 + (rr0 + uc)];
      }
    }
    __syncthreads();

    // ---- online softmax over this 64-wide key tile ----
    {
      int r = tid & 63, ch = tid >> 6;
      float mx = -3.0e38f;
      for (int c = ch * 16; c < ch * 16 + 16; ++c) mx = fmaxf(mx, sS[r * 64 + c]);
      red[r * 4 + ch] = mx;
    }
    __syncthreads();
    if (tid < 64) {
      float mx = fmaxf(fmaxf(red[tid * 4 + 0], red[tid * 4 + 1]),
                       fmaxf(red[tid * 4 + 2], red[tid * 4 + 3]));
      float mo = mrow[tid];
      float mn = fmaxf(mo, mx);
      float al = __expf(mo - mn);
      mrow[tid] = mn; arow[tid] = al; lrow[tid] *= al;
    }
    __syncthreads();
    {
      int r = tid & 63, ch = tid >> 6;
      float mn = mrow[r];
      float sm = 0.0f;
      for (int c = ch * 16; c < ch * 16 + 16; ++c) {
        float p = __expf(sS[r * 64 + c] - mn);
        Ps[r * 64 + c] = (bf16)p;
        sm += p;
      }
      red[r * 4 + ch] = sm;
    }
    __syncthreads();
    if (tid < 64)
      lrow[tid] += red[tid * 4 + 0] + red[tid * 4 + 1] + red[tid * 4 + 2] + red[tid * 4 + 3];
#pragma unroll
    for (int ct = 0; ct < 2; ++ct)               // rescale running O by alpha(row)
#pragma unroll
      for (int g = 0; g < 8; ++g) {
        int rl = 16 * wr + g + lh * 8;
        oacc[ct][g] *= arow[rl];
      }
    // ---- O += P @ V ----
    v16bf pa0 = ldA(Ps, 64, 16 * wr, 0, lane);
    v16bf pa1 = ldA(Ps, 64, 16 * wr, 32, lane);
#pragma unroll
    for (int ct = 0; ct < 2; ++ct) {
      v16bf b0 = ldB(vs, 64, 32 * wc + 16 * ct, 0, lane);
      v16bf b1 = ldB(vs, 64, 32 * wc + 16 * ct, 32, lane);
      oacc[ct] = wmma_bf16(pa0, b0, oacc[ct]);
      oacc[ct] = wmma_bf16(pa1, b1, oacc[ct]);
    }
  }
  __syncthreads();
#pragma unroll
  for (int ct = 0; ct < 2; ++ct)                 // write attn out in (t, n, e) layout
#pragma unroll
    for (int g = 0; g < 8; ++g) {
      int rl = 16 * wr + g + lh * 8;
      int cl = 32 * wc + 16 * ct + ln;
      O[(size_t)(i0 + rl) * 8192 + headoff + cl] = oacc[ct][g] / lrow[rl];
    }
}

// -------------------- launch -----------------------------------------------
extern "C" void kernel_launch(void* const* d_in, const int* in_sizes, int n_in,
                              void* d_out, int out_size, void* d_ws, size_t ws_size,
                              hipStream_t stream) {
  (void)in_sizes; (void)n_in; (void)out_size; (void)ws_size;
  const float* q_src = (const float*)d_in[0];   // (512, 8, 1024)
  const float* k_src = (const float*)d_in[1];   // (512, 8, 1024)
  const float* v_src = (const float*)d_in[2];   // (512, 8, 1024)
  const float* mask  = (const float*)d_in[3];   // (8, 512, 512)
  const float* rf    = (const float*)d_in[4];   // (8, 1025, 64) -> (8200, 64)
  const float* Wq    = (const float*)d_in[6];   // (1024, 1024)
  const float* Wk    = (const float*)d_in[7];
  const float* Wkl   = (const float*)d_in[8];   // (1024, 64)
  const float* Wv    = (const float*)d_in[9];
  const float* Wo    = (const float*)d_in[10];
  const float* qcb   = (const float*)d_in[11];  // (1, 1024)
  const float* qlb   = (const float*)d_in[12];  // (1, 1024)

  float* ws = (float*)d_ws;
  float* Qp  = ws;                    // 4096x1024
  float* Kp  = Qp  + 4194304;         // 4096x1024
  float* Vp  = Kp  + 4194304;         // 4096x1024
  float* LKp = Vp  + 4194304;         // 8200x1024
  float* cbw = LKp + 8396800;         // 128x512
  float* ubw = cbw + 65536;           // 128x1025
  float* AO  = ubw + 131200;          // 4096x1024

  dim3 blk(256);
  gemm_bt_kernel<false><<<dim3(8, 32), blk, 0, stream>>>(q_src, Wq, Qp, 4096, 1024, 1024);
  gemm_bt_kernel<false><<<dim3(8, 32), blk, 0, stream>>>(k_src, Wk, Kp, 4096, 1024, 1024);
  gemm_bt_kernel<false><<<dim3(8, 32), blk, 0, stream>>>(v_src, Wv, Vp, 4096, 1024, 1024);
  gemm_bt_kernel<true ><<<dim3(8, 65), blk, 0, stream>>>(rf, Wkl, LKp, 8200, 1024, 64);
  cbias_kernel<<<256, 256, 0, stream>>>(Kp, qcb, cbw);
  ubias_kernel<<<513, 256, 0, stream>>>(LKp, qlb, ubw);
  attn_kernel<<<dim3(128, 8), blk, 0, stream>>>(Qp, Kp, Vp, LKp, cbw, ubw, mask, AO);
  gemm_bt_kernel<false><<<dim3(8, 32), blk, 0, stream>>>(AO, Wo, (float*)d_out, 4096, 1024, 1024);
}
